// GAT_54520314856149
// MI455X (gfx1250) — matrix operand order repaired and checked by reading, not compiled, not run on hardware
//
#include <hip/hip_runtime.h>
#include <hip/hip_bf16.h>

// ---------------------------------------------------------------------------
// GAT (4x GATConv + MLP head) for gfx1250 (MI455X).
//  - Dense GEMMs use v_wmma_f32_16x16x32_bf16 (bf16 in, fp32 accum).
//  - Edge softmax/aggregation via L2-resident atomics (h fits in 192MB L2).
// ---------------------------------------------------------------------------

typedef __attribute__((ext_vector_type(16))) __bf16 v16bf;
typedef __attribute__((ext_vector_type(8)))  __bf16 v8bf;
typedef __attribute__((ext_vector_type(8)))  float  v8f;
typedef __hip_bfloat16 bf16_t;

__device__ __forceinline__ float  bf2f(bf16_t v) { return __bfloat162float(v); }
__device__ __forceinline__ bf16_t f2bf(float v)  { return __float2bfloat16(v); }

// Order-preserving float -> uint mapping (for atomic segment max).
__device__ __forceinline__ unsigned fmap(float f) {
    unsigned u = __float_as_uint(f);
    return (u & 0x80000000u) ? ~u : (u | 0x80000000u);
}
__device__ __forceinline__ float funmap(unsigned u) {
    unsigned v = (u & 0x80000000u) ? (u & 0x7FFFFFFFu) : ~u;
    return __uint_as_float(v);
}

// ---------------------------------------------------------------------------
// bf16 GEMM: C[M,Ncols] (f32) = A[M,K] (bf16 row-major) * B, with B given
// pre-transposed as BT[Ncols,K] (bf16 row-major). One wave per 16x64 C strip.
// Requires: M % 16 == 0, Ncols % 64 == 0, K % 32 == 0.
// ---------------------------------------------------------------------------
__global__ __launch_bounds__(256) void gat_gemm_bf16(
    const bf16_t* __restrict__ A, const bf16_t* __restrict__ BT,
    float* __restrict__ C, int M, int Ncols, int K)
{
    const int lane = threadIdx.x & 31;
    const int wave = threadIdx.x >> 5;
    const int tiles_n = Ncols >> 6;
    const int tiles   = (M >> 4) * tiles_n;
    const int tile = blockIdx.x * 8 + wave;
    if (tile >= tiles) return;                       // wave-uniform exit
    const int tm = tile / tiles_n;
    const int tn = tile - tm * tiles_n;
    const int hf  = lane >> 4;                       // half-wave: 0 or 1
    const int l16 = lane & 15;

    // A fragment: lane holds row (tm*16+l16); halves 0/1 own K 0-15 / 16-31
    const __bf16* Ap = (const __bf16*)A + (size_t)(tm * 16 + l16) * K + hf * 16;
    // B fragment (from BT): lane holds column (tn*64 + sub*16 + l16);
    // halves own K 0-15 / 16-31 (contiguous along K in BT rows).
    const __bf16* B0 = (const __bf16*)BT + (size_t)(tn * 64 + l16) * K + hf * 16;
    const size_t bstep = (size_t)16 * K;

    v8f c0 = {}, c1 = {}, c2 = {}, c3 = {};
    for (int k = 0; k < K; k += 32) {
        v8bf a0 = *(const v8bf*)(Ap + k);
        v8bf a1 = *(const v8bf*)(Ap + k + 8);
        v16bf a = __builtin_shufflevector(a0, a1,
                    0,1,2,3,4,5,6,7,8,9,10,11,12,13,14,15);
        v16bf b0 = *(const v16bf*)(B0 + k);
        v16bf b1 = *(const v16bf*)(B0 + bstep + k);
        v16bf b2 = *(const v16bf*)(B0 + 2 * bstep + k);
        v16bf b3 = *(const v16bf*)(B0 + 3 * bstep + k);
        c0 = __builtin_amdgcn_wmma_f32_16x16x32_bf16(false, a, false, b0, (short)0, c0, false, false);
        c1 = __builtin_amdgcn_wmma_f32_16x16x32_bf16(false, a, false, b1, (short)0, c1, false, false);
        c2 = __builtin_amdgcn_wmma_f32_16x16x32_bf16(false, a, false, b2, (short)0, c2, false, false);
        c3 = __builtin_amdgcn_wmma_f32_16x16x32_bf16(false, a, false, b3, (short)0, c3, false, false);
    }
    // C layout: VGPR r -> row r (lanes 0-15) / row 8+r (lanes 16-31), col = lane%16
    float* Cp = C + (size_t)(tm * 16 + hf * 8) * Ncols + tn * 64 + l16;
    for (int r = 0; r < 8; ++r) {
        Cp[0]  = c0[r];
        Cp[16] = c1[r];
        Cp[32] = c2[r];
        Cp[48] = c3[r];
        Cp += Ncols;
    }
}

// Transpose + zero-pad weight: W[K,Nc] f32 -> WT[Nc,Kpad] bf16.
__global__ void wtrans_k(const float* __restrict__ W, bf16_t* __restrict__ WT,
                         int K, int Nc, int Kpad)
{
    int i = blockIdx.x * blockDim.x + threadIdx.x;
    if (i >= Nc * Kpad) return;
    int nc = i / Kpad, k = i - nc * Kpad;
    float v = (k < K) ? W[(size_t)k * Nc + nc] : 0.f;
    WT[i] = f2bf(v);
}

// x[N,10] f32 -> xb[N,32] bf16 zero-padded.
__global__ void xpad_k(const float* __restrict__ x, bf16_t* __restrict__ xb, int n)
{
    int i = blockIdx.x * blockDim.x + threadIdx.x;
    if (i >= n * 32) return;
    int node = i >> 5, k = i & 31;
    float v = (k < 10) ? x[(size_t)node * 10 + k] : 0.f;
    xb[i] = f2bf(v);
}

// alpha_s / alpha_d dots per (node, head).
__global__ void node_alpha_k(const float* __restrict__ h,
                             const float* __restrict__ a_s, const float* __restrict__ a_d,
                             float* __restrict__ als, float* __restrict__ ald,
                             int n, int Hh, int C)
{
    int i = blockIdx.x * blockDim.x + threadIdx.x;
    if (i >= n * Hh) return;
    int node = i / Hh, hd = i - node * Hh;
    const float* hp  = h + ((size_t)node * Hh + hd) * C;
    const float* asp = a_s + (size_t)hd * C;
    const float* adp = a_d + (size_t)hd * C;
    float s = 0.f, d = 0.f;
    for (int c = 0; c < C; ++c) { float v = hp[c]; s += v * asp[c]; d += v * adp[c]; }
    als[i] = s; ald[i] = d;
}

// Per-layer init: agg=0, amax=mapped(-inf floor)=0, denom=0.
__global__ void layer_init_k(float* __restrict__ agg, unsigned* __restrict__ amaxu,
                             float* __restrict__ denom, int nHC, int nH)
{
    int i = blockIdx.x * blockDim.x + threadIdx.x;
    if (i < nHC) agg[i] = 0.f;
    if (i < nH) { amaxu[i] = 0u; denom[i] = 0.f; }
}

// Per (edge, head): leaky-relu(alpha_s[src]+alpha_d[dst]); segment-max into dst.
__global__ void edge_alpha_k(const int* __restrict__ ei, int E, int Etot,
                             const float* __restrict__ als, const float* __restrict__ ald,
                             float* __restrict__ alpha, unsigned* __restrict__ amaxu, int Hh)
{
    int i = blockIdx.x * blockDim.x + threadIdx.x;
    if (i >= Etot * Hh) return;
    int e = i / Hh, hd = i - e * Hh;
    int s = (e < E) ? ei[e] : (e - E);
    int d = (e < E) ? ei[E + e] : (e - E);
    float a = als[(size_t)s * Hh + hd] + ald[(size_t)d * Hh + hd];
    a = (a > 0.f) ? a : 0.2f * a;
    alpha[i] = a;
    atomicMax(amaxu + (size_t)d * Hh + hd, fmap(a));
}

// exp(alpha - amax[dst]); segment-sum denom[dst].
__global__ void edge_exp_k(const int* __restrict__ ei, int E, int Etot,
                           const float* __restrict__ alpha, const unsigned* __restrict__ amaxu,
                           float* __restrict__ eexp, float* __restrict__ denom, int Hh)
{
    int i = blockIdx.x * blockDim.x + threadIdx.x;
    if (i >= Etot * Hh) return;
    int e = i / Hh, hd = i - e * Hh;
    int d = (e < E) ? ei[E + e] : (e - E);
    float m = funmap(amaxu[(size_t)d * Hh + hd]);
    float ev = __expf(alpha[i] - m);
    eexp[i] = ev;
    atomicAdd(denom + (size_t)d * Hh + hd, ev);
}

// agg[dst] += h[src] * coef  (float4 per thread; gather + atomic scatter in L2).
__global__ void edge_agg_k(const int* __restrict__ ei, int E, int Etot,
                           const float* __restrict__ h, const float* __restrict__ eexp,
                           const float* __restrict__ denom, float* __restrict__ agg,
                           int Hh, int C, int HC)
{
    long long i = (long long)blockIdx.x * blockDim.x + threadIdx.x;
    int C4 = C >> 2;
    long long tot = (long long)Etot * Hh * C4;
    if (i >= tot) return;
    int c4 = (int)(i % C4);
    long long t = i / C4;
    int hd = (int)(t % Hh);
    int e  = (int)(t / Hh);
    int s = (e < E) ? ei[e] : (e - E);
    int d = (e < E) ? ei[E + e] : (e - E);
    float coef = eexp[(size_t)e * Hh + hd] /
                 (denom[(size_t)d * Hh + hd] + 1e-16f);
    const float4 hv = *(const float4*)(h + (size_t)s * HC + hd * C + c4 * 4);
    float* op = agg + (size_t)d * HC + hd * C + c4 * 4;
    atomicAdd(op + 0, hv.x * coef);
    atomicAdd(op + 1, hv.y * coef);
    atomicAdd(op + 2, hv.z * coef);
    atomicAdd(op + 3, hv.w * coef);
}

// bias + ELU -> bf16 (next GEMM input).
__global__ void fin_gat_k(const float* __restrict__ agg, const float* __restrict__ bias,
                          bf16_t* __restrict__ outb, int total, int HC)
{
    int i = blockIdx.x * blockDim.x + threadIdx.x;
    if (i >= total) return;
    float v = agg[i] + bias[i % HC];
    v = (v > 0.f) ? v : (__expf(v) - 1.f);
    outb[i] = f2bf(v);
}

// bias + ReLU -> bf16 (MLP).
__global__ void fin_relu_k(const float* __restrict__ hin, const float* __restrict__ bias,
                           bf16_t* __restrict__ outb, int total, int HC)
{
    int i = blockIdx.x * blockDim.x + threadIdx.x;
    if (i >= total) return;
    float v = hin[i] + bias[i % HC];
    outb[i] = f2bf(v > 0.f ? v : 0.f);
}

// Final 256->1 dot + sigmoid.
__global__ void final_k(const bf16_t* __restrict__ m2, const float* __restrict__ Wm3,
                        const float* __restrict__ bm3, float* __restrict__ out, int n)
{
    int i = blockIdx.x * blockDim.x + threadIdx.x;
    if (i >= n) return;
    const bf16_t* row = m2 + (size_t)i * 256;
    float acc = bm3[0];
    for (int j = 0; j < 256; ++j) acc += bf2f(row[j]) * Wm3[j];
    out[i] = 1.f / (1.f + __expf(-acc));
}

// ---------------------------------------------------------------------------
extern "C" void kernel_launch(void* const* d_in, const int* in_sizes, int n_in,
                              void* d_out, int out_size, void* d_ws, size_t ws_size,
                              hipStream_t stream)
{
    const float* x    = (const float*)d_in[0];
    const int*   ei   = (const int*)d_in[1];
    const float* W1   = (const float*)d_in[2];
    const float* as1  = (const float*)d_in[3];
    const float* ad1  = (const float*)d_in[4];
    const float* b1   = (const float*)d_in[5];
    const float* W2   = (const float*)d_in[6];
    const float* as2  = (const float*)d_in[7];
    const float* ad2  = (const float*)d_in[8];
    const float* b2   = (const float*)d_in[9];
    const float* W3   = (const float*)d_in[10];
    const float* as3  = (const float*)d_in[11];
    const float* ad3  = (const float*)d_in[12];
    const float* b3   = (const float*)d_in[13];
    const float* W4   = (const float*)d_in[14];
    const float* as4  = (const float*)d_in[15];
    const float* ad4  = (const float*)d_in[16];
    const float* b4   = (const float*)d_in[17];
    const float* Wm1  = (const float*)d_in[18];
    const float* bm1  = (const float*)d_in[19];
    const float* Wm2  = (const float*)d_in[20];
    const float* bm2  = (const float*)d_in[21];
    const float* Wm3  = (const float*)d_in[22];
    const float* bm3  = (const float*)d_in[23];

    const int n    = in_sizes[0] / 10;   // 50000 (divisible by 16)
    const int E    = in_sizes[1] / 2;    // 800000
    const int Etot = E + n;              // with self-loops

    // ---- carve workspace ----
    char* base = (char*)d_ws;
    size_t off = 0;
    auto take = [&](size_t bytes) -> void* {
        off = (off + 255) & ~(size_t)255;
        void* p = base + off;
        off += bytes;
        return p;
    };
    bf16_t* xb    = (bf16_t*)take((size_t)n * 32 * 2);
    bf16_t* w1t   = (bf16_t*)take((size_t)256 * 32 * 2);
    bf16_t* w2t   = (bf16_t*)take((size_t)512 * 256 * 2);
    bf16_t* w3t   = (bf16_t*)take((size_t)512 * 512 * 2);
    bf16_t* w4t   = (bf16_t*)take((size_t)64 * 512 * 2);
    bf16_t* wm1t  = (bf16_t*)take((size_t)256 * 64 * 2);
    bf16_t* wm2t  = (bf16_t*)take((size_t)256 * 256 * 2);
    float*  hbuf  = (float*)take((size_t)n * 512 * 4);
    float*  agg   = (float*)take((size_t)n * 512 * 4);
    bf16_t* actb  = (bf16_t*)take((size_t)n * 512 * 2);
    float*  als   = (float*)take((size_t)n * 4 * 4);
    float*  ald   = (float*)take((size_t)n * 4 * 4);
    float*  alpha = (float*)take((size_t)Etot * 4 * 4);
    float*  eexp  = (float*)take((size_t)Etot * 4 * 4);
    unsigned* amaxu = (unsigned*)take((size_t)n * 4 * 4);
    float*  denom = (float*)take((size_t)n * 4 * 4);
    (void)ws_size; (void)n_in; (void)out_size;

    auto cdiv = [](long long a, long long b) -> int { return (int)((a + b - 1) / b); };

    // ---- one-time-per-call weight/input conversions ----
    wtrans_k<<<cdiv(256 * 32, 256), 256, 0, stream>>>(W1, w1t, 10, 256, 32);
    wtrans_k<<<cdiv(512 * 256, 256), 256, 0, stream>>>(W2, w2t, 256, 512, 256);
    wtrans_k<<<cdiv(512 * 512, 256), 256, 0, stream>>>(W3, w3t, 512, 512, 512);
    wtrans_k<<<cdiv(64 * 512, 256), 256, 0, stream>>>(W4, w4t, 512, 64, 512);
    wtrans_k<<<cdiv(256 * 64, 256), 256, 0, stream>>>(Wm1, wm1t, 64, 256, 64);
    wtrans_k<<<cdiv(256 * 256, 256), 256, 0, stream>>>(Wm2, wm2t, 256, 256, 256);
    xpad_k<<<cdiv((long long)n * 32, 256), 256, 0, stream>>>(x, xb, n);

    // ---- one GAT layer ----
    auto run_gat = [&](const bf16_t* inA, int K, const bf16_t* WT, int HC, int Hh,
                       const float* a_s, const float* a_d, const float* bias) {
        const int C = HC / Hh;
        const int tiles = (n / 16) * (HC / 64);
        gat_gemm_bf16<<<cdiv(tiles, 8), 256, 0, stream>>>(inA, WT, hbuf, n, HC, K);
        const int nh = n * Hh;
        node_alpha_k<<<cdiv(nh, 256), 256, 0, stream>>>(hbuf, a_s, a_d, als, ald, n, Hh, C);
        const int nHC = n * HC;
        layer_init_k<<<cdiv(nHC, 256), 256, 0, stream>>>(agg, amaxu, denom, nHC, nh);
        const int eh = Etot * Hh;
        edge_alpha_k<<<cdiv(eh, 256), 256, 0, stream>>>(ei, E, Etot, als, ald, alpha, amaxu, Hh);
        edge_exp_k<<<cdiv(eh, 256), 256, 0, stream>>>(ei, E, Etot, alpha, amaxu, eexp, denom, Hh);
        const long long tot = (long long)Etot * Hh * (C / 4);
        edge_agg_k<<<cdiv(tot, 256), 256, 0, stream>>>(ei, E, Etot, hbuf, eexp, denom, agg,
                                                       Hh, C, HC);
        fin_gat_k<<<cdiv(nHC, 256), 256, 0, stream>>>(agg, bias, actb, nHC, HC);
    };

    run_gat(xb,   32,  w1t, 256, 4, as1, ad1, b1);   // 10(->32) -> 4x64
    run_gat(actb, 256, w2t, 512, 4, as2, ad2, b2);   // 256 -> 4x128
    run_gat(actb, 512, w3t, 512, 4, as3, ad3, b3);   // 512 -> 4x128
    run_gat(actb, 512, w4t, 64,  1, as4, ad4, b4);   // 512 -> 1x64

    // ---- MLP head ----
    {
        int tiles = (n / 16) * (256 / 64);
        gat_gemm_bf16<<<cdiv(tiles, 8), 256, 0, stream>>>(actb, wm1t, hbuf, n, 256, 64);
        fin_relu_k<<<cdiv((long long)n * 256, 256), 256, 0, stream>>>(hbuf, bm1, actb, n * 256, 256);
        gat_gemm_bf16<<<cdiv(tiles, 8), 256, 0, stream>>>(actb, wm2t, hbuf, n, 256, 256);
        fin_relu_k<<<cdiv((long long)n * 256, 256), 256, 0, stream>>>(hbuf, bm2, actb, n * 256, 256);
        final_k<<<cdiv(n, 256), 256, 0, stream>>>(actb, Wm3, bm3, (float*)d_out, n);
    }
}
